// RNNAttnDecoder_62491774156954
// MI455X (gfx1250) — compile-verified, matrix operand order checked by reading, not run on hardware
//
#include <hip/hip_runtime.h>
#include <math.h>

// Problem dims
#define V_  32000
#define E_  512
#define H_  1024
#define L_  2
#define B_  64
#define S_  2048
#define R_  (B_ * S_)   // 131072 flattened (b,s) rows

typedef __attribute__((ext_vector_type(8)))  float   v8f;
typedef __attribute__((ext_vector_type(16))) __bf16  v16bf;
typedef __attribute__((ext_vector_type(4)))  float   f4;

union Frag {
    uint4          u[2];   // 32 bytes
    unsigned short s[16];
    v16bf          v;
};

// float -> bf16 bits, round-to-nearest-even
__device__ __forceinline__ unsigned short f2bf(float f) {
    unsigned int u = __float_as_uint(f);
    u += 0x7FFFu + ((u >> 16) & 1u);
    return (unsigned short)(u >> 16);
}

// ISA 7.12.2 16-bit A-matrix layout: element (row-in-tile mm, kk in 0..31)
// lane = mm + 16*bit3(kk), half = (kk&7) + 8*bit4(kk). B mirrors with nn.
__device__ __forceinline__ int frag_lane(int mn, int kk) {
    return mn + (((kk >> 3) & 1) << 4);
}
__device__ __forceinline__ int frag_half(int kk) {
    return (kk & 7) + ((kk >> 4) << 3);
}

__device__ __forceinline__ void loadA_lds(Frag& a, const unsigned short* sA,
                                          int mt, int ks, int lane) {
    const uint4* ap = (const uint4*)(sA + ((((mt * 32 + ks) * 32) + lane) << 4));
    a.u[0] = ap[0]; a.u[1] = ap[1];
}
__device__ __forceinline__ void loadB4(Frag* b, const unsigned short* __restrict__ Uap,
                                       int nt0, int ks, int lane) {
#pragma unroll
    for (int j = 0; j < 4; ++j) {
        const uint4* bp = (const uint4*)(Uap + (((((nt0 + j) * 32 + ks) * 32) + lane) << 4));
        b[j].u[0] = bp[0]; b[j].u[1] = bp[1];
    }
}

// ---------------------------------------------------------------------------
// P1: pack Ua (H x H f32, row k, col n) into per-fragment bf16 layout:
//     [n_tile(64)][k_step(32)][lane(32)][half(16)]
// ---------------------------------------------------------------------------
__global__ __launch_bounds__(256) void pack_Ua_k(const float* __restrict__ Ua,
                                                 unsigned short* __restrict__ Uap) {
    int idx = blockIdx.x * 256 + threadIdx.x;        // 0 .. H*H-1
    int k = idx >> 10, n = idx & (H_ - 1);
    int nt = n >> 4, nn = n & 15, ks = k >> 5, kk = k & 31;
    Uap[((((nt * 32 + ks) * 32) + frag_lane(nn, kk)) << 4) + frag_half(kk)] = f2bf(Ua[idx]);
}

// ---------------------------------------------------------------------------
// P0: q = hidden[1] @ Wa   (64 x 1024). b-tiled x8 so hidden reads are
// wave-uniform (scalar loads) and Wa rows are coalesced.
// ---------------------------------------------------------------------------
__global__ __launch_bounds__(256) void q_gemm_k(const float* __restrict__ hidden,
                                                const float* __restrict__ Wa,
                                                float* __restrict__ q) {
    int t = threadIdx.x;
    int h = blockIdx.x * 256 + t;
    int b0 = blockIdx.y * 8;
    float acc[8] = {0.f, 0.f, 0.f, 0.f, 0.f, 0.f, 0.f, 0.f};
    const float* hid1 = hidden + (size_t)B_ * H_;    // layer L-1 = 1
    for (int k = 0; k < H_; ++k) {
        float w = Wa[(size_t)k * H_ + h];
#pragma unroll
        for (int j = 0; j < 8; ++j) acc[j] += hid1[(size_t)(b0 + j) * H_ + k] * w;
    }
#pragma unroll
    for (int j = 0; j < 8; ++j) q[(b0 + j) * H_ + h] = acc[j];
}

// ---------------------------------------------------------------------------
// K2 (dominant): fused  score[b,s] = sum_n v[n]*tanh(q[b,n] + (enc@Ua)[b,s,n])
// Block = 128 rows (one b), 512 thr = 16 waves = 8 M-tiles x 2 N-halves.
// A fragments staged in 256KB dynamic LDS (bf16, packed b64 stores); B frags
// double-buffered from the packed-Ua L2-resident buffer so global loads for
// k+1 overlap the WMMAs for k.
// ---------------------------------------------------------------------------
#define SMEM_A_U16 (8 * 32 * 32 * 16)                // 131072 u16 = 256 KB

__global__ __launch_bounds__(512) void score_gemm_k(const float* __restrict__ enc,
                                                    const unsigned short* __restrict__ Uap,
                                                    const float* __restrict__ q,
                                                    const float* __restrict__ vvec,
                                                    float* __restrict__ score) {
    extern __shared__ unsigned char smemRaw[];
    unsigned short* sA = (unsigned short*)smemRaw;
    float* sScore = (float*)(smemRaw + (size_t)SMEM_A_U16 * 2);

    const int t = threadIdx.x;
    const int base = blockIdx.x * 128;               // flattened row base
    const int b = base >> 11;                        // base / S

    // Stage 128 rows x 1024 K of enc as bf16 A-fragments.
    // float4 NT loads (enc is single-use streaming); 4 consecutive K fall in
    // 4 consecutive fragment halves -> one packed 8-byte LDS store.
    const f4* enc4 = (const f4*)(enc + (size_t)base * H_);
    for (int i = 0; i < 64; ++i) {
        int e = i * 512 + t;                         // f4-group index 0..32767
        int m = e >> 8;                              // row in block 0..127
        int g = e & 255;                             // f4 group in row
        f4 f = __builtin_nontemporal_load(enc4 + (size_t)m * (H_ / 4) + g);
        int k = g * 4;
        int mt = m >> 4, mm = m & 15, ks = k >> 5, kk = k & 31;
        unsigned int lo = (unsigned int)f2bf(f.x) | ((unsigned int)f2bf(f.y) << 16);
        unsigned int hi2 = (unsigned int)f2bf(f.z) | ((unsigned int)f2bf(f.w) << 16);
        uint2 p; p.x = lo; p.y = hi2;
        *(uint2*)&sA[((((mt * 32 + ks) * 32) + frag_lane(mm, kk)) << 4) + frag_half(kk)] = p;
    }
    if (t < 128) sScore[t] = 0.f;
    __syncthreads();

    const int wave = t >> 5, lane = t & 31;
    const int mt = wave & 7;                         // M-tile 0..7
    const int nhalf = wave >> 3;                     // N half 0..1
    const int nn = lane & 15, hi = lane >> 4;

    for (int ng = 0; ng < 8; ++ng) {
        int nt0 = nhalf * 32 + ng * 4;
        v8f c[4] = {};
        Frag bA[4], bB[4];
        loadB4(bA, Uap, nt0, 0, lane);
        for (int ks = 0; ks < 32; ks += 2) {
            Frag a0; loadA_lds(a0, sA, mt, ks, lane);
            loadB4(bB, Uap, nt0, ks + 1, lane);
#pragma unroll
            for (int j = 0; j < 4; ++j)
                c[j] = __builtin_amdgcn_wmma_f32_16x16x32_bf16(
                    false, a0.v, false, bA[j].v, (short)0, c[j], false, false);
            Frag a1; loadA_lds(a1, sA, mt, ks + 1, lane);
            // ks==30 prefetches fragment-step 32: a 1KB overread landing in the
            // next workspace region (wsQ) -- allocated memory, values unused.
            loadB4(bA, Uap, nt0, ks + 2, lane);
#pragma unroll
            for (int j = 0; j < 4; ++j)
                c[j] = __builtin_amdgcn_wmma_f32_16x16x32_bf16(
                    false, a1.v, false, bB[j].v, (short)0, c[j], false, false);
        }
        // Epilogue: v[n]*tanh(q+k), reduce across the 16 N lanes per M row.
#pragma unroll
        for (int j = 0; j < 4; ++j) {
            int n = (nt0 + j) * 16 + nn;
            float qv = q[b * H_ + n];
            float vv = vvec[n];
#pragma unroll
            for (int r = 0; r < 8; ++r) {
                float x = vv * tanhf(qv + c[j][r]);
                x += __shfl_xor(x, 1, 16);
                x += __shfl_xor(x, 2, 16);
                x += __shfl_xor(x, 4, 16);
                x += __shfl_xor(x, 8, 16);
                if (nn == 0) atomicAdd(&sScore[mt * 16 + r + hi * 8], x);  // ds_add_f32
            }
        }
    }
    __syncthreads();
    if (t < 128) score[base + t] = sScore[t];
}

// ---------------------------------------------------------------------------
// K3: row softmax over S=2048 per batch element
// ---------------------------------------------------------------------------
__global__ __launch_bounds__(256) void softmax_k(const float* __restrict__ score,
                                                 float* __restrict__ w) {
    __shared__ float red[256];
    int b = blockIdx.x, t = threadIdx.x;
    const float* s = score + (size_t)b * S_;
    float mx = -1e30f;
    for (int i = t; i < S_; i += 256) mx = fmaxf(mx, s[i]);
    red[t] = mx; __syncthreads();
    for (int st = 128; st > 0; st >>= 1) { if (t < st) red[t] = fmaxf(red[t], red[t + st]); __syncthreads(); }
    mx = red[0]; __syncthreads();
    float sum = 0.f;
    for (int i = t; i < S_; i += 256) { float e = __expf(s[i] - mx); w[(size_t)b * S_ + i] = e; sum += e; }
    red[t] = sum; __syncthreads();
    for (int st = 128; st > 0; st >>= 1) { if (t < st) red[t] += red[t + st]; __syncthreads(); }
    float inv = 1.f / red[0];
    for (int i = t; i < S_; i += 256) w[(size_t)b * S_ + i] *= inv;
}

// ---------------------------------------------------------------------------
// K4: context = weights @ enc  (second streaming pass over enc, NT float4)
// Partials over S-chunks for parallelism, then reduce.
// ---------------------------------------------------------------------------
__global__ __launch_bounds__(256) void ctx_part_k(const float* __restrict__ enc,
                                                  const float* __restrict__ w,
                                                  float* __restrict__ part) {
    int b = blockIdx.x, chunk = blockIdx.y, t = threadIdx.x;
    __shared__ float sw[256];
    sw[t] = w[(size_t)b * S_ + chunk * 256 + t];
    __syncthreads();
    const f4* e4 = (const f4*)(enc + ((size_t)b * S_ + (size_t)chunk * 256) * H_) + t;
    f4 acc = {0.f, 0.f, 0.f, 0.f};
    for (int s = 0; s < 256; ++s) {
        f4 e = __builtin_nontemporal_load(e4 + (size_t)s * (H_ / 4));
        acc += sw[s] * e;
    }
    float* p = part + ((size_t)chunk * B_ + b) * H_ + t * 4;
    p[0] = acc.x; p[1] = acc.y; p[2] = acc.z; p[3] = acc.w;
}

__global__ __launch_bounds__(256) void ctx_reduce_k(const float* __restrict__ part,
                                                    float* __restrict__ ctx) {
    int idx = blockIdx.x * 256 + threadIdx.x;        // 0 .. B*H-1
    float a = 0.f;
#pragma unroll
    for (int c = 0; c < 8; ++c) a += part[(size_t)c * B_ * H_ + idx];
    ctx[idx] = a;
}

// ---------------------------------------------------------------------------
// K5: h0 = tanh([emb, ctx] @ Wih0 + bih0 + hidden[0] @ Whh0)
// b-tiled x8; x-operands are wave-uniform -> scalar loads; weights coalesced.
// ---------------------------------------------------------------------------
__global__ __launch_bounds__(256) void h0_k(const int* __restrict__ tok,
                                            const float* __restrict__ embt,
                                            const float* __restrict__ ctx,
                                            const float* __restrict__ hidden,
                                            const float* __restrict__ Wih0,
                                            const float* __restrict__ bih0,
                                            const float* __restrict__ Whh0,
                                            float* __restrict__ h0out) {
    int t = threadIdx.x;
    int h = blockIdx.x * 256 + t;
    int b0 = blockIdx.y * 8;
    const float* er[8];
#pragma unroll
    for (int j = 0; j < 8; ++j) er[j] = embt + (size_t)tok[b0 + j] * E_;
    float acc[8];
#pragma unroll
    for (int j = 0; j < 8; ++j) acc[j] = bih0[h];
    for (int k = 0; k < E_; ++k) {
        float w = Wih0[(size_t)k * H_ + h];
#pragma unroll
        for (int j = 0; j < 8; ++j) acc[j] += er[j][k] * w;
    }
    for (int k = 0; k < H_; ++k) {
        float w = Wih0[(size_t)(E_ + k) * H_ + h];
#pragma unroll
        for (int j = 0; j < 8; ++j) acc[j] += ctx[(b0 + j) * H_ + k] * w;
    }
    for (int k = 0; k < H_; ++k) {
        float w = Whh0[(size_t)k * H_ + h];
#pragma unroll
        for (int j = 0; j < 8; ++j) acc[j] += hidden[(size_t)(b0 + j) * H_ + k] * w;
    }
#pragma unroll
    for (int j = 0; j < 8; ++j) h0out[(b0 + j) * H_ + h] = tanhf(acc[j]);
}

// K6: h1 = tanh(h0 @ Wih1 + bih1 + hidden[1] @ Whh1)
__global__ __launch_bounds__(256) void h1_k(const float* __restrict__ h0,
                                            const float* __restrict__ hidden,
                                            const float* __restrict__ Wih1,
                                            const float* __restrict__ bih1,
                                            const float* __restrict__ Whh1,
                                            float* __restrict__ h1out) {
    int t = threadIdx.x;
    int h = blockIdx.x * 256 + t;
    int b0 = blockIdx.y * 8;
    const float* hid1 = hidden + (size_t)B_ * H_;
    float acc[8];
#pragma unroll
    for (int j = 0; j < 8; ++j) acc[j] = bih1[h];
    for (int k = 0; k < H_; ++k) {
        float w = Wih1[(size_t)k * H_ + h];
#pragma unroll
        for (int j = 0; j < 8; ++j) acc[j] += h0[(b0 + j) * H_ + k] * w;
    }
    for (int k = 0; k < H_; ++k) {
        float w = Whh1[(size_t)k * H_ + h];
#pragma unroll
        for (int j = 0; j < 8; ++j) acc[j] += hid1[(size_t)(b0 + j) * H_ + k] * w;
    }
#pragma unroll
    for (int j = 0; j < 8; ++j) h1out[(b0 + j) * H_ + h] = tanhf(acc[j]);
}

// P2: pack [h1, ctx] (64 x 2048) into bf16 A-fragment layout for logits GEMM
__global__ __launch_bounds__(256) void pack_xA_k(const float* __restrict__ h1,
                                                 const float* __restrict__ ctx,
                                                 unsigned short* __restrict__ Ap) {
    int idx = blockIdx.x * 256 + threadIdx.x;        // 0 .. 64*2048-1
    int m = idx >> 11, k = idx & 2047;
    float x = (k < H_) ? h1[m * H_ + k] : ctx[m * H_ + (k - H_)];
    int mt = m >> 4, mm = m & 15, ks = k >> 5, kk = k & 31;
    Ap[((((mt * 64 + ks) * 32) + frag_lane(mm, kk)) << 4) + frag_half(kk)] = f2bf(x);
}

// ---------------------------------------------------------------------------
// K7: logits = [h1,ctx](64x2048) @ Wout(2048x32000) + bout.  Memory bound on
// Wout: one N-tile per wave (32 cols/block -> 1000 blocks for occupancy),
// NT-stream Wout f32 into registers as one clause, convert to bf16, WMMA.
// global_prefetch_b8 readahead for the next K slab.
// ---------------------------------------------------------------------------
__global__ __launch_bounds__(256) void logits_gemm_k(const unsigned short* __restrict__ Ap,
                                                     const float* __restrict__ Wout,
                                                     const float* __restrict__ bout,
                                                     float* __restrict__ out) {
    int t = threadIdx.x, wave = t >> 5, lane = t & 31;
    int mt = wave & 3;
    int ng = wave >> 2;
    int nt = blockIdx.x * 2 + ng;
    int n0 = nt << 4;
    int nn = lane & 15, hi = lane >> 4;
    int kkBase = hi << 3;
    v8f c = {};
    for (int ks = 0; ks < 64; ++ks) {
        float raw[16];
#pragma unroll
        for (int h = 0; h < 16; ++h) {
            int kk = (h & 7) + kkBase + ((h >> 3) << 4);
            raw[h] = __builtin_nontemporal_load(Wout + (size_t)(ks * 32 + kk) * V_ + n0 + nn);
        }
        Frag a;
        const uint4* ap = (const uint4*)(Ap + ((((mt * 64 + ks) * 32) + lane) << 4));
        a.u[0] = ap[0]; a.u[1] = ap[1];
        // speculative readahead of next K slab (dropped if OOB on last iter)
        __builtin_prefetch(Wout + (size_t)((ks + 1) * 32 + kkBase) * V_ + n0 + nn, 0, 0);
        Frag bf;
#pragma unroll
        for (int h = 0; h < 16; ++h) bf.s[h] = f2bf(raw[h]);
        c = __builtin_amdgcn_wmma_f32_16x16x32_bf16(
            false, a.v, false, bf.v, (short)0, c, false, false);
    }
    float bb = bout[n0 + nn];
#pragma unroll
    for (int r = 0; r < 8; ++r) {
        int m = mt * 16 + r + hi * 8;
        out[(size_t)m * V_ + n0 + nn] = c[r] + bb;
    }
}

// ---------------------------------------------------------------------------
extern "C" void kernel_launch(void* const* d_in, const int* in_sizes, int n_in,
                              void* d_out, int out_size, void* d_ws, size_t ws_size,
                              hipStream_t stream) {
    const int*   tok    = (const int*)  d_in[0];
    const float* hidden = (const float*)d_in[1];
    const float* enc    = (const float*)d_in[2];
    const float* embt   = (const float*)d_in[3];
    const float* Wa     = (const float*)d_in[4];
    const float* Ua     = (const float*)d_in[5];
    const float* vvec   = (const float*)d_in[6];
    const float* Wih0   = (const float*)d_in[7];
    const float* bih0   = (const float*)d_in[8];
    const float* Whh0   = (const float*)d_in[9];
    const float* Wih1   = (const float*)d_in[10];
    const float* bih1   = (const float*)d_in[11];
    const float* Whh1   = (const float*)d_in[12];
    const float* Wout   = (const float*)d_in[13];
    const float* bout   = (const float*)d_in[14];

    float* out      = (float*)d_out;
    float* out_h0   = out + (size_t)B_ * V_;                 // new_hidden[0]
    float* out_h1   = out_h0 + (size_t)B_ * H_;              // new_hidden[1]

    // Workspace partition (~5.75 MB total)
    char* ws = (char*)d_ws;
    unsigned short* wsUap   = (unsigned short*)(ws);                       // 2 MB
    float*          wsQ     = (float*)(ws + (2u << 20));                   // 256 KB
    float*          wsScore = (float*)(ws + (2u << 20) + (256u << 10));    // 512 KB
    float*          wsWgt   = (float*)(ws + (2u << 20) + (768u << 10));    // 512 KB
    float*          wsPart  = (float*)(ws + (2u << 20) + (1280u << 10));   // 2 MB
    float*          wsCtx   = (float*)(ws + (4u << 20) + (1280u << 10));   // 256 KB
    unsigned short* wsXA    = (unsigned short*)(ws + (4u << 20) + (1536u << 10)); // 256 KB

    // P1: pack Ua to bf16 fragments (L2-resident B matrix)
    pack_Ua_k<<<(H_ * H_) / 256, 256, 0, stream>>>(Ua, wsUap);
    // P0: q = hidden[1] @ Wa
    q_gemm_k<<<dim3(H_ / 256, B_ / 8), 256, 0, stream>>>(hidden, Wa, wsQ);
    // K2: fused enc@Ua + tanh + v-dot  (dominant WMMA kernel)
    score_gemm_k<<<R_ / 128, 512, SMEM_A_U16 * 2 + 128 * 4, stream>>>(enc, wsUap, wsQ, vvec, wsScore);
    // K3: softmax
    softmax_k<<<B_, 256, 0, stream>>>(wsScore, wsWgt);
    // K4: context (second enc pass)
    ctx_part_k<<<dim3(B_, 8), 256, 0, stream>>>(enc, wsWgt, wsPart);
    ctx_reduce_k<<<(B_ * H_) / 256, 256, 0, stream>>>(wsPart, wsCtx);
    // K5/K6: recurrent layers -> new_hidden
    h0_k<<<dim3(H_ / 256, B_ / 8), 256, 0, stream>>>(tok, embt, wsCtx, hidden, Wih0, bih0, Whh0, out_h0);
    h1_k<<<dim3(H_ / 256, B_ / 8), 256, 0, stream>>>(out_h0, hidden, Wih1, bih1, Whh1, out_h1);
    // P2 + K7: logits WMMA
    pack_xA_k<<<(B_ * 2048) / 256, 256, 0, stream>>>(out_h1, wsCtx, wsXA);
    logits_gemm_k<<<V_ / 32, 256, 0, stream>>>(wsXA, Wout, bout, out);
}
